// MultiHeadLatentAttention_81844896792614
// MI455X (gfx1250) — matrix operand order repaired and checked by reading, not compiled, not run on hardware
//
#include <hip/hip_runtime.h>
#include <cstdint>
#include <cstddef>

// ---------- types ----------
typedef __attribute__((ext_vector_type(16))) __bf16 v16bf;
typedef __attribute__((ext_vector_type(8)))  float  v8f;

typedef int v4i __attribute__((vector_size(16)));
typedef v4i __attribute__((address_space(1)))* gv4i_t;   // global ptr to int4
typedef v4i __attribute__((address_space(3)))* lv4i_t;   // LDS ptr to int4

#if defined(__HIP_DEVICE_COMPILE__) && defined(__gfx1250__) && \
    __has_builtin(__builtin_amdgcn_global_load_async_to_lds_b128) && \
    __has_builtin(__builtin_amdgcn_s_wait_asynccnt)
#define USE_ASYNC_LDS 1
#else
#define USE_ASYNC_LDS 0
#endif

// ---------- problem constants ----------
#define BATCH     2
#define SEQ       2048
#define HIDDEN    4096
#define NUM_HEADS 4
#define HEAD_DIM  1024
#define LATENT    1024
#define ATT_SCALE 0.03125f   // 1024^-0.5

// ---------- GEMM tile config ----------
#define BM 128
#define BN 128
#define BK 32
#define LDSS 40              // padded LDS stride (bf16); 80B rows, 16B aligned
#define GEMM_THREADS 256     // 8 waves of 32

union Frag16 { uint4 u[2]; v16bf v; };

// C = alpha * A(MxK,row) @ B^T where B stored (NxK,row) ("NT" GEMM).
// Output f32 or bf16. M%128==0, N%128==0, K%32==0 required.
template <bool OUT_BF16>
__global__ __launch_bounds__(GEMM_THREADS)
void gemm_nt_bf16_wmma(const __bf16* __restrict__ A, const __bf16* __restrict__ Bm,
                       void* __restrict__ Cv,
                       int M, int N, int K, int lda, int ldb, int ldc, float alpha)
{
    __shared__ __align__(16) __bf16 As[2][BM * LDSS];
    __shared__ __align__(16) __bf16 Bs[2][BN * LDSS];

    const int tid   = threadIdx.x;
    const int lane  = tid & 31;
    const int wave  = tid >> 5;        // 0..7
    const int wm    = wave >> 1;       // 0..3 -> 32 rows each
    const int wn    = wave & 1;        // 0..1 -> 64 cols each
    const int lmod  = lane & 15;
    const int lhalf = lane >> 4;       // 0/1

    const int m0 = blockIdx.y * BM;
    const int n0 = blockIdx.x * BN;

    // staging: 256 threads, each thread owns 16 bf16 (32B) of the 128x32 tile
    const int sr = tid >> 1;           // 0..127 (row of tile)
    const int sk = (tid & 1) * 16;     // 0 / 16

    const __bf16* gA = A  + (size_t)(m0 + sr) * lda + sk;
    const __bf16* gB = Bm + (size_t)(n0 + sr) * ldb + sk;
    const int sOff = sr * LDSS + sk;

    v8f vzero = {};
    v8f acc[2][4];
    #pragma unroll
    for (int mt = 0; mt < 2; ++mt)
        #pragma unroll
        for (int nt = 0; nt < 4; ++nt) acc[mt][nt] = vzero;

    auto stage = [&](int buf, int k0) {
#if USE_ASYNC_LDS
        // async copy 32B of A and 32B of B per thread; imm offset applies to
        // both the global and the LDS address (ISA 15.18.3)
        __builtin_amdgcn_global_load_async_to_lds_b128(
            (gv4i_t)(gA + k0), (lv4i_t)(&As[buf][sOff]), 0, 0);
        __builtin_amdgcn_global_load_async_to_lds_b128(
            (gv4i_t)(gA + k0), (lv4i_t)(&As[buf][sOff]), 16, 0);
        __builtin_amdgcn_global_load_async_to_lds_b128(
            (gv4i_t)(gB + k0), (lv4i_t)(&Bs[buf][sOff]), 0, 0);
        __builtin_amdgcn_global_load_async_to_lds_b128(
            (gv4i_t)(gB + k0), (lv4i_t)(&Bs[buf][sOff]), 16, 0);
#else
        {
            const uint4* g = reinterpret_cast<const uint4*>(gA + k0);
            uint4 x0 = g[0], x1 = g[1];
            uint4* s = reinterpret_cast<uint4*>(&As[buf][sOff]);
            s[0] = x0; s[1] = x1;
        }
        {
            const uint4* g = reinterpret_cast<const uint4*>(gB + k0);
            uint4 x0 = g[0], x1 = g[1];
            uint4* s = reinterpret_cast<uint4*>(&Bs[buf][sOff]);
            s[0] = x0; s[1] = x1;
        }
        if (k0 + BK < K) {
            __builtin_prefetch(gA + k0 + BK, 0, 3);
            __builtin_prefetch(gB + k0 + BK, 0, 3);
        }
#endif
    };

    auto compute = [&](int buf) {
        Frag16 fa[2], fb[4];
        // A frag (16x32): lanes 0-15 -> K 0..7 & 16..23 ; lanes 16-31 -> 8..15 & 24..31
        #pragma unroll
        for (int mt = 0; mt < 2; ++mt) {
            const __bf16* p = &As[buf][(wm * 32 + mt * 16 + lmod) * LDSS + lhalf * 8];
            fa[mt].u[0] = *reinterpret_cast<const uint4*>(p);
            fa[mt].u[1] = *reinterpret_cast<const uint4*>(p + 16);
        }
        // B frag (32x16): lane n = lmod, lanes 0-15 -> K 0..7 & 16..23 ; lanes 16-31 -> 8..15 & 24..31
        #pragma unroll
        for (int nt = 0; nt < 4; ++nt) {
            const __bf16* p = &Bs[buf][(wn * 64 + nt * 16 + lmod) * LDSS + lhalf * 8];
            fb[nt].u[0] = *reinterpret_cast<const uint4*>(p);
            fb[nt].u[1] = *reinterpret_cast<const uint4*>(p + 16);
        }
        #pragma unroll
        for (int mt = 0; mt < 2; ++mt)
            #pragma unroll
            for (int nt = 0; nt < 4; ++nt)
                acc[mt][nt] = __builtin_amdgcn_wmma_f32_16x16x32_bf16(
                    false, fa[mt].v, false, fb[nt].v, (short)0, acc[mt][nt], false, false);
    };

#if USE_ASYNC_LDS
    // double-buffered: overlap async tile copy with WMMA on the other buffer
    stage(0, 0);
    __builtin_amdgcn_s_wait_asynccnt(0);
    __syncthreads();
    int buf = 0;
    for (int k0 = 0; k0 < K; k0 += BK) {
        if (k0 + BK < K) stage(buf ^ 1, k0 + BK);
        compute(buf);
        __builtin_amdgcn_s_wait_asynccnt(0);
        __syncthreads();
        buf ^= 1;
    }
#else
    for (int k0 = 0; k0 < K; k0 += BK) {
        __syncthreads();
        stage(0, k0);
        __syncthreads();
        compute(0);
    }
#endif

    // writeback: D layout: VGPR r -> M = tile_m + lhalf*8 + r ; N = tile_n + lmod
    #pragma unroll
    for (int mt = 0; mt < 2; ++mt) {
        #pragma unroll
        for (int nt = 0; nt < 4; ++nt) {
            const int gr = m0 + wm * 32 + mt * 16 + lhalf * 8;
            const int gc = n0 + wn * 64 + nt * 16 + lmod;
            #pragma unroll
            for (int r = 0; r < 8; ++r) {
                float vv = alpha * acc[mt][nt][r];
                if (OUT_BF16)
                    reinterpret_cast<__bf16*>(Cv)[(size_t)(gr + r) * ldc + gc] = (__bf16)vv;
                else
                    reinterpret_cast<float*>(Cv)[(size_t)(gr + r) * ldc + gc] = vv;
            }
        }
    }
}

// ---------- elementwise kernels ----------
__global__ void cvt_f32_bf16(const float* __restrict__ x, __bf16* __restrict__ y, size_t n4)
{
    size_t i = (size_t)blockIdx.x * blockDim.x + threadIdx.x;
    if (i >= n4) return;
    const float4 v = reinterpret_cast<const float4*>(x)[i];
    __bf16* o = y + i * 4;
    o[0] = (__bf16)v.x; o[1] = (__bf16)v.y; o[2] = (__bf16)v.z; o[3] = (__bf16)v.w;
}

// y (C x R, bf16) = transpose(x (R x C, f32)); R,C multiples of 32
__global__ __launch_bounds__(256)
void transpose_cvt(const float* __restrict__ x, __bf16* __restrict__ y, int R, int C)
{
    __shared__ float tile[32][33];
    const int bx = blockIdx.x * 32;   // col base in x
    const int by = blockIdx.y * 32;   // row base in x
    const int tx = threadIdx.x;       // 0..31
    const int ty = threadIdx.y;       // 0..7
    #pragma unroll
    for (int i = ty; i < 32; i += 8)
        tile[i][tx] = x[(size_t)(by + i) * C + (bx + tx)];
    __syncthreads();
    #pragma unroll
    for (int i = ty; i < 32; i += 8)
        y[(size_t)(bx + i) * R + (by + tx)] = (__bf16)tile[tx][i];
}

// RoPE (interleaved pairs) + f32 -> bf16
__global__ void rope_to_bf16(const float* __restrict__ x, __bf16* __restrict__ y,
                             int rows, int cols, int head_dim, int seqlen)
{
    size_t idx = (size_t)blockIdx.x * blockDim.x + threadIdx.x;
    size_t pairs_per_row = (size_t)cols / 2;
    size_t npairs = (size_t)rows * pairs_per_row;
    if (idx >= npairs) return;
    size_t row = idx / pairs_per_row;
    int    c0  = (int)(idx % pairs_per_row) * 2;
    int    pos = (int)(row % (size_t)seqlen);
    int    d   = c0 % head_dim;                       // = 2*i
    float  freq = __powf(10000.0f, -(float)d / (float)head_dim);
    float  ang  = (float)pos * freq;
    float  s, c;
    __sincosf(ang, &s, &c);
    const float* xr = x + row * (size_t)cols + c0;
    float x0 = xr[0], x1 = xr[1];
    __bf16* yr = y + row * (size_t)cols + c0;
    yr[0] = (__bf16)(x0 * c - x1 * s);
    yr[1] = (__bf16)(x1 * c + x0 * s);
}

// row softmax over S columns with additive mask; f32 in -> bf16 probs out
__global__ __launch_bounds__(256)
void softmax_rows(const float* __restrict__ scores, const float* __restrict__ mask,
                  __bf16* __restrict__ probs, int S)
{
    const int row = blockIdx.x;
    const float* sr = scores + (size_t)row * S;
    const float* mr = mask   + (size_t)row * S;
    __shared__ float red[256];

    float lmax = -3.0e38f;
    for (int j = threadIdx.x; j < S; j += 256)
        lmax = fmaxf(lmax, sr[j] + mr[j]);
    red[threadIdx.x] = lmax; __syncthreads();
    for (int off = 128; off > 0; off >>= 1) {
        if (threadIdx.x < off)
            red[threadIdx.x] = fmaxf(red[threadIdx.x], red[threadIdx.x + off]);
        __syncthreads();
    }
    const float m = red[0]; __syncthreads();

    float lsum = 0.0f;
    for (int j = threadIdx.x; j < S; j += 256)
        lsum += __expf(sr[j] + mr[j] - m);
    red[threadIdx.x] = lsum; __syncthreads();
    for (int off = 128; off > 0; off >>= 1) {
        if (threadIdx.x < off)
            red[threadIdx.x] += red[threadIdx.x + off];
        __syncthreads();
    }
    const float inv = 1.0f / red[0];

    __bf16* pr = probs + (size_t)row * S;
    for (int j = threadIdx.x; j < S; j += 256)
        pr[j] = (__bf16)(__expf(sr[j] + mr[j] - m) * inv);
}

// ---------- host-side helpers ----------
static void launch_gemm_nt(const __bf16* A, const __bf16* Bt, void* C,
                           int M, int N, int K, int lda, int ldb, int ldc,
                           bool out_bf16, float alpha, hipStream_t s)
{
    dim3 grid(N / BN, M / BM), block(GEMM_THREADS);
    if (out_bf16)
        gemm_nt_bf16_wmma<true ><<<grid, block, 0, s>>>(A, Bt, C, M, N, K, lda, ldb, ldc, alpha);
    else
        gemm_nt_bf16_wmma<false><<<grid, block, 0, s>>>(A, Bt, C, M, N, K, lda, ldb, ldc, alpha);
}

static void launch_cvt(const float* x, __bf16* y, size_t n, hipStream_t s)
{
    size_t n4 = n / 4;
    int grid = (int)((n4 + 255) / 256);
    cvt_f32_bf16<<<grid, 256, 0, s>>>(x, y, n4);
}

static void launch_transpose_cvt(const float* x, __bf16* y, int R, int C, hipStream_t s)
{
    dim3 grid(C / 32, R / 32), block(32, 8);
    transpose_cvt<<<grid, block, 0, s>>>(x, y, R, C);
}

extern "C" void kernel_launch(void* const* d_in, const int* in_sizes, int n_in,
                              void* d_out, int out_size, void* d_ws, size_t ws_size,
                              hipStream_t stream)
{
    (void)in_sizes; (void)n_in; (void)out_size; (void)ws_size;
    const float* hs   = (const float*)d_in[0];   // (B,S,4096)
    const float* mask = (const float*)d_in[1];   // (1,1,S,S)
    const float* Wq   = (const float*)d_in[2];   // (4096,4096)
    const float* Wk   = (const float*)d_in[3];   // (4096,1024)
    const float* Wv   = (const float*)d_in[4];   // (4096,1024)
    const float* Wo   = (const float*)d_in[5];   // (4096,4096)
    float*       out  = (float*)d_out;           // (B,S,4096)

    const size_t ROWS = (size_t)BATCH * SEQ;     // 4096

    // workspace carve-up
    char* p = (char*)d_ws;
    __bf16* hs_bf = (__bf16*)p;  p += ROWS * HIDDEN * 2;                  // 32MB
    __bf16* wqT   = (__bf16*)p;  p += (size_t)HIDDEN * HIDDEN * 2;        // 32MB (4096x4096)
    __bf16* wkT   = (__bf16*)p;  p += (size_t)HIDDEN * LATENT * 2;        // 8MB  (1024x4096)
    __bf16* wvT   = (__bf16*)p;  p += (size_t)HIDDEN * LATENT * 2;        // 8MB  (1024x4096)
    __bf16* woT   = (__bf16*)p;  p += (size_t)HIDDEN * HIDDEN * 2;        // 32MB (4096x4096)
    float*  q_f32 = (float*)p;   p += ROWS * HIDDEN * 4;                  // 64MB
    float*  k_f32 = (float*)p;   p += ROWS * LATENT * 4;                  // 16MB
    float*  v_f32 = (float*)p;   p += ROWS * LATENT * 4;                  // 16MB
    __bf16* q_bf  = (__bf16*)p;  p += ROWS * HIDDEN * 2;                  // 32MB
    __bf16* k_bf  = (__bf16*)p;  p += ROWS * LATENT * 2;                  // 8MB
    __bf16* vT_bf = (__bf16*)p;  p += ROWS * LATENT * 2;                  // 8MB  (1024x4096)
    __bf16* attn_bf = (__bf16*)p; /* p += ROWS * HIDDEN * 2; */           // 32MB
    // fp32 scratch reused once dead:
    float*  scores = q_f32;                  // 16MB needed (<= 64MB)
    __bf16* probs  = (__bf16*)k_f32;         // 8MB needed (<= 16MB)

    // 1) precision conversion; all GEMM B operands stored N x K (transposed)
    launch_cvt(hs, hs_bf, ROWS * HIDDEN, stream);
    launch_transpose_cvt(Wq, wqT, HIDDEN, HIDDEN, stream);   // -> (4096 x 4096), ld 4096
    launch_transpose_cvt(Wk, wkT, HIDDEN, LATENT, stream);   // -> (1024 x 4096), ld 4096
    launch_transpose_cvt(Wv, wvT, HIDDEN, LATENT, stream);   // -> (1024 x 4096), ld 4096
    launch_transpose_cvt(Wo, woT, HIDDEN, HIDDEN, stream);   // -> (4096 x 4096), ld 4096

    // 2) projections (WMMA bf16 NT, f32 out)
    launch_gemm_nt(hs_bf, wqT, q_f32, (int)ROWS, HIDDEN, HIDDEN, HIDDEN, HIDDEN, HIDDEN, false, 1.0f, stream);
    launch_gemm_nt(hs_bf, wkT, k_f32, (int)ROWS, LATENT, HIDDEN, HIDDEN, HIDDEN, LATENT, false, 1.0f, stream);
    launch_gemm_nt(hs_bf, wvT, v_f32, (int)ROWS, LATENT, HIDDEN, HIDDEN, HIDDEN, LATENT, false, 1.0f, stream);

    // 3) RoPE(q,k) -> bf16 ; v -> bf16 transposed (1024 x 4096, d-major)
    {
        size_t npairs_q = ROWS * HIDDEN / 2;
        rope_to_bf16<<<(int)((npairs_q + 255) / 256), 256, 0, stream>>>(q_f32, q_bf, (int)ROWS, HIDDEN, HEAD_DIM, SEQ);
        size_t npairs_k = ROWS * LATENT / 2;
        rope_to_bf16<<<(int)((npairs_k + 255) / 256), 256, 0, stream>>>(k_f32, k_bf, (int)ROWS, LATENT, HEAD_DIM, SEQ);
        launch_transpose_cvt(v_f32, vT_bf, (int)ROWS, LATENT, stream);
    }

    // 4) attention per (b, h): scores -> softmax -> p@v
    for (int b = 0; b < BATCH; ++b) {
        const __bf16* kb  = k_bf  + (size_t)b * SEQ * LATENT;   // (2048 keys x 1024), NT-ready
        const __bf16* vTb = vT_bf + (size_t)b * SEQ;            // cols b*S.. of (1024 x 4096)
        for (int h = 0; h < NUM_HEADS; ++h) {
            const __bf16* qh = q_bf + (size_t)b * SEQ * HIDDEN + (size_t)h * HEAD_DIM;
            __bf16* ah = attn_bf + (size_t)b * SEQ * HIDDEN + (size_t)h * HEAD_DIM;
            // scores = scale * q @ k^T
            launch_gemm_nt(qh, kb, scores, SEQ, SEQ, HEAD_DIM,
                           HIDDEN, LATENT, SEQ, false, ATT_SCALE, stream);
            // softmax(+causal mask) -> bf16 probs
            softmax_rows<<<SEQ, 256, 0, stream>>>(scores, mask, probs, SEQ);
            // attn = probs @ v ; B operand is v^T slice (N=1024 x K=2048, ld 4096)
            launch_gemm_nt(probs, vTb, ah, SEQ, HEAD_DIM, SEQ,
                           SEQ, HIDDEN, HIDDEN, true, 1.0f, stream);
        }
    }

    // 5) out = attn @ Wo (f32 out)
    launch_gemm_nt(attn_bf, woT, out, (int)ROWS, HIDDEN, HIDDEN,
                   HIDDEN, HIDDEN, HIDDEN, false, 1.0f, stream);
}
